// GNNEncoder_33071248179561
// MI455X (gfx1250) — matrix operand order repaired and checked by reading, not compiled, not run on hardware
//
#include <hip/hip_runtime.h>
#include <hip/hip_bf16.h>
#include <math.h>

typedef __attribute__((ext_vector_type(16))) _Float16 v16h;
typedef __attribute__((ext_vector_type(8)))  float    v8f;

#define HID 64
#define BATCHES 500
#define BN_EPS 1e-5f

__device__ __forceinline__ float softplus_f(float z) {
    // numerically stable log(1+exp(z))
    return fmaxf(z, 0.f) + log1pf(expf(-fabsf(z)));
}

// ---------------------------------------------------------------------------
// utility kernels
// ---------------------------------------------------------------------------
__global__ void zero_kernel(float* p, long n) {
    for (long i = (long)blockIdx.x * blockDim.x + threadIdx.x; i < n;
         i += (long)gridDim.x * blockDim.x)
        p[i] = 0.f;
}

__global__ void copy_kernel(float* dst, const float* src, long n) {
    for (long i = (long)blockIdx.x * blockDim.x + threadIdx.x; i < n;
         i += (long)gridDim.x * blockDim.x)
        dst[i] = src[i];
}

// sum and sum-of-squares of x[0..E) -> xstats[0], xstats[1]
__global__ void reduce_x_kernel(const float* __restrict__ x, float* xstats, int E) {
    __shared__ float l1[256], l2[256];
    int t = threadIdx.x;
    float s = 0.f, s2 = 0.f;
    for (long i = (long)blockIdx.x * 256 + t; i < E; i += (long)gridDim.x * 256) {
        float v = x[i];
        s += v; s2 += v * v;
    }
    l1[t] = s; l2[t] = s2;
    __syncthreads();
    for (int off = 128; off > 0; off >>= 1) {
        if (t < off) { l1[t] += l1[t + off]; l2[t] += l2[t + off]; }
        __syncthreads();
    }
    if (t == 0) { atomicAdd(&xstats[0], l1[0]); atomicAdd(&xstats[1], l2[0]); }
}

// ---------------------------------------------------------------------------
// fused: d_act = softplus(BN1(x*w_d1+b_d1)) computed in LDS (f16),
// d2_pre = d_act @ w_d2^T + b_d2 via v_wmma_f32_16x16x32_f16 (K=64 as 2 steps),
// store d2_pre as f16 and accumulate BN2 per-column sum/sumsq via atomics.
// Block: 256 threads = 8 wave32s; covers 32 edge rows x 64 cols
// (8 tiles of 16x16, one per wave).
// ---------------------------------------------------------------------------
__global__ void gemm_d2_kernel(const float* __restrict__ x,
                               const float* __restrict__ w_d1,
                               const float* __restrict__ g_d1,
                               const float* __restrict__ be_d1,
                               const float* __restrict__ w_d2,
                               const float* __restrict__ b_d2,
                               const float* __restrict__ xstats,
                               float* __restrict__ d2stats,
                               _Float16* __restrict__ d2_pre,
                               int E) {
    __shared__ _Float16 lB[64 * 64];   // B[k][n] = w_d2[n][k]  (f16)
    __shared__ _Float16 lA[32 * 64];   // 32 rows of activations (f16)
    __shared__ float sScale[64], sBeta[64];
    __shared__ float sSum[64], sSumSq[64];

    int t = threadIdx.x;
    for (int i = t; i < 64 * 64; i += 256) {
        int k = i >> 6, n = i & 63;
        lB[i] = (_Float16)w_d2[n * 64 + k];
    }
    float invE = 1.f / (float)E;
    float mx = xstats[0] * invE;
    float vx = xstats[1] * invE - mx * mx;
    if (t < 64) {
        // BN1 of rank-1 layer factorizes: scale_h = w1*rsqrt(vx*w1^2+eps)*g1
        float w = w_d1[t];
        sScale[t] = w * rsqrtf(vx * w * w + BN_EPS) * g_d1[t];
        sBeta[t]  = be_d1[t];          // b_d1 cancels inside BN
        sSum[t] = 0.f; sSumSq[t] = 0.f;
    }
    __syncthreads();

    int rowBase = blockIdx.x * 32;
    for (int i = t; i < 32 * 64; i += 256) {
        int r = i >> 6, h = i & 63;
        int e = rowBase + r;
        float a = 0.f;
        if (e < E) a = softplus_f((x[e] - mx) * sScale[h] + sBeta[h]);
        lA[i] = (_Float16)a;
    }
    __syncthreads();

    int wave = t >> 5, lane = t & 31;
    int rt = wave >> 2, ct = wave & 3;      // row-tile 0..1, col-tile 0..3
    int halfid = lane >> 4;
    int mrow = rt * 16 + (lane & 15);

    // A fragments (16x32 f16): VGPR v: grp=v>>2, k0 = grp*16 + half*8 + (v&3)*2
    v16h a0, a1, b0, b1;
#pragma unroll
    for (int v = 0; v < 8; ++v) {
        int k0 = (v >> 2) * 16 + halfid * 8 + (v & 3) * 2;
        a0[2 * v]     = lA[mrow * 64 + k0];
        a0[2 * v + 1] = lA[mrow * 64 + k0 + 1];
        a1[2 * v]     = lA[mrow * 64 + 32 + k0];
        a1[2 * v + 1] = lA[mrow * 64 + 32 + k0 + 1];
    }
    // B fragments (32x16 f16): same K order as A: k = grp*16 + half*8 + (v&3)*2
    int ncol = ct * 16 + (lane & 15);
#pragma unroll
    for (int v = 0; v < 8; ++v) {
        int k = (v >> 2) * 16 + halfid * 8 + (v & 3) * 2;
        b0[2 * v]     = lB[k * 64 + ncol];
        b0[2 * v + 1] = lB[(k + 1) * 64 + ncol];
        b1[2 * v]     = lB[(32 + k) * 64 + ncol];
        b1[2 * v + 1] = lB[(33 + k) * 64 + ncol];
    }

    v8f c = {};
    c = __builtin_amdgcn_wmma_f32_16x16x32_f16(false, a0, false, b0,
                                               (short)0, c, false, false);
    c = __builtin_amdgcn_wmma_f32_16x16x32_f16(false, a1, false, b1,
                                               (short)0, c, false, false);

    // D layout: VGPR r -> M = rt*16 + 8*half + r, N = ncol
    float bias = b_d2[ncol];
    float lsum = 0.f, lsq = 0.f;
#pragma unroll
    for (int r = 0; r < 8; ++r) {
        int m = rt * 16 + 8 * halfid + r;
        int e = rowBase + m;
        if (e < E) {
            float dv = c[r] + bias;
            d2_pre[(size_t)e * 64 + ncol] = (_Float16)dv;
            lsum += dv; lsq += dv * dv;
        }
    }
    atomicAdd(&sSum[ncol], lsum);
    atomicAdd(&sSumSq[ncol], lsq);
    __syncthreads();
    if (t < 64) {
        atomicAdd(&d2stats[t], sSum[t]);
        atomicAdd(&d2stats[64 + t], sSumSq[t]);
    }
}

// ---------------------------------------------------------------------------
// ea = BN2(d2_pre) * edge_emb[edge_type]   (in-place on the f16 buffer)
// ---------------------------------------------------------------------------
__global__ void bn2_ea_kernel(_Float16* __restrict__ d2ea,
                              const float* __restrict__ d2stats,
                              const float* __restrict__ g,
                              const float* __restrict__ be,
                              const float* __restrict__ edge_emb,
                              const int* __restrict__ edge_type,
                              int E) {
    __shared__ float sM[64], sR[64], sG[64], sB[64];
    int t = threadIdx.x;
    if (t < 64) {
        float invE = 1.f / (float)E;
        float m = d2stats[t] * invE;
        float v = d2stats[64 + t] * invE - m * m;
        sM[t] = m; sR[t] = rsqrtf(v + BN_EPS);
        sG[t] = g[t]; sB[t] = be[t];
    }
    __syncthreads();
    size_t total = (size_t)E * 16;   // 4 features per work item
    for (size_t i = (size_t)blockIdx.x * 256 + t; i < total;
         i += (size_t)gridDim.x * 256) {
        size_t e = i >> 4;
        int h0 = (int)(i & 15) * 4;
        int et = edge_type[e];
#pragma unroll
        for (int j = 0; j < 4; ++j) {
            int h = h0 + j;
            float d = (float)d2ea[e * 64 + h];
            float val = (d - sM[h]) * sR[h] * sG[h] + sB[h];
            d2ea[e * 64 + h] = (_Float16)(val * edge_emb[et * 64 + h]);
        }
    }
}

// h0 = node_emb[node_type]
__global__ void gather_nodes_kernel(const float* __restrict__ node_emb,
                                    const int* __restrict__ node_type,
                                    float* __restrict__ h, int N) {
    size_t total = (size_t)N * 64;
    for (size_t i = (size_t)blockIdx.x * 256 + threadIdx.x; i < total;
         i += (size_t)gridDim.x * 256) {
        size_t n = i >> 6;
        h[i] = node_emb[(size_t)node_type[n] * 64 + (i & 63)];
    }
}

// agg += softplus(h[src] + ea)   (agg pre-initialized to h -> "+ h" term)
__global__ void gconv_scatter_kernel(const float* __restrict__ h,
                                     const _Float16* __restrict__ ea,
                                     const int* __restrict__ src,
                                     const int* __restrict__ dst,
                                     float* __restrict__ agg, int E) {
    size_t total = (size_t)E * 16;
    for (size_t i = (size_t)blockIdx.x * 256 + threadIdx.x; i < total;
         i += (size_t)gridDim.x * 256) {
        size_t e = i >> 4;
        int h0 = (int)(i & 15) * 4;
        int s = src[e], d = dst[e];
#pragma unroll
        for (int j = 0; j < 4; ++j) {
            int hh = h0 + j;
            float z = h[(size_t)s * 64 + hh] + (float)ea[e * 64 + hh];
            atomicAdd(&agg[(size_t)d * 64 + hh], softplus_f(z));
        }
    }
}

// per-column sum / sumsq over [rows, C]; C in {32, 64}; stats has 2C floats
__global__ void colstats_kernel(const float* __restrict__ in, float* stats,
                                int rows, int C) {
    __shared__ float l1[256], l2[256];
    int t = threadIdx.x;
    int c = t & (C - 1);
    int rl = t / C;
    int rpt = 256 / C;
    float s = 0.f, s2 = 0.f;
    for (long r = (long)blockIdx.x * rpt + rl; r < rows;
         r += (long)gridDim.x * rpt) {
        float v = in[r * (long)C + c];
        s += v; s2 += v * v;
    }
    l1[t] = s; l2[t] = s2;
    __syncthreads();
    if (t < C) {
        for (int k = 1; k < rpt; ++k) { s += l1[c + k * C]; s2 += l2[c + k * C]; }
        atomicAdd(&stats[c], s);
        atomicAdd(&stats[C + c], s2);
    }
}

// out = BN(in) (optionally softplus); elementwise over [rows, C]
__global__ void bn_apply_kernel(const float* __restrict__ in,
                                float* __restrict__ out,
                                const float* __restrict__ stats,
                                const float* __restrict__ g,
                                const float* __restrict__ be,
                                int rows, int C, int act) {
    long total = (long)rows * C;
    float invN = 1.f / (float)rows;
    for (long i = (long)blockIdx.x * 256 + threadIdx.x; i < total;
         i += (long)gridDim.x * 256) {
        int c = (int)(i & (C - 1));
        float m = stats[c] * invN;
        float v = stats[C + c] * invN - m * m;
        float z = (in[i] - m) * rsqrtf(v + BN_EPS) * g[c] + be[c];
        if (act) z = softplus_f(z);
        out[i] = z;
    }
}

// h_global[batch[n]] += h[n]
__global__ void hg_scatter_kernel(const float* __restrict__ h,
                                  const int* __restrict__ batch,
                                  float* __restrict__ hg, int N) {
    size_t total = (size_t)N * 16;
    for (size_t i = (size_t)blockIdx.x * 256 + threadIdx.x; i < total;
         i += (size_t)gridDim.x * 256) {
        size_t n = i >> 4;
        int h0 = (int)(i & 15) * 4;
        int b = batch[n];
#pragma unroll
        for (int j = 0; j < 4; ++j)
            atomicAdd(&hg[(size_t)b * 64 + h0 + j], h[n * 64 + h0 + j]);
    }
}

// o1_pre = [h, hg[batch]] @ w_o1^T + b_o1 ; w_o1 is [64,128]
__global__ void o1_kernel(const float* __restrict__ h,
                          const float* __restrict__ hg,
                          const int* __restrict__ batch,
                          const float* __restrict__ w,
                          const float* __restrict__ b,
                          float* __restrict__ out, int N) {
    __shared__ float lw[64 * 128];
    __shared__ float lx[4][128];
    int t = threadIdx.x;
    for (int i = t; i < 64 * 128; i += 256) lw[i] = w[i];
    __syncthreads();
    for (int nb = blockIdx.x * 4; nb < N; nb += gridDim.x * 4) {
        __syncthreads();
        for (int i = t; i < 4 * 128; i += 256) {
            int r = i >> 7, k = i & 127;
            int n = nb + r;
            float v = 0.f;
            if (n < N)
                v = (k < 64) ? h[(size_t)n * 64 + k]
                             : hg[(size_t)batch[n] * 64 + (k - 64)];
            lx[r][k] = v;
        }
        __syncthreads();
        int r = t >> 6, ho = t & 63;
        int n = nb + r;
        if (n < N) {
            float acc = b[ho];
            for (int k = 0; k < 128; ++k) acc += lx[r][k] * lw[ho * 128 + k];
            out[(size_t)n * 64 + ho] = acc;
        }
    }
}

// o2_pre = in @ w_o2^T + b_o2 ; w_o2 is [32,64]
__global__ void o2_kernel(const float* __restrict__ in,
                          const float* __restrict__ w,
                          const float* __restrict__ b,
                          float* __restrict__ out, int N) {
    __shared__ float lw[32 * 64];
    __shared__ float lx[8][64];
    int t = threadIdx.x;
    for (int i = t; i < 32 * 64; i += 256) lw[i] = w[i];
    __syncthreads();
    for (int nb = blockIdx.x * 8; nb < N; nb += gridDim.x * 8) {
        __syncthreads();
        for (int i = t; i < 8 * 64; i += 256) {
            int r = i >> 6, k = i & 63;
            int n = nb + r;
            lx[r][k] = (n < N) ? in[(size_t)n * 64 + k] : 0.f;
        }
        __syncthreads();
        int r = t >> 5, ho = t & 31;
        int n = nb + r;
        if (n < N) {
            float acc = b[ho];
            for (int k = 0; k < 64; ++k) acc += lx[r][k] * lw[ho * 64 + k];
            out[(size_t)n * 32 + ho] = acc;
        }
    }
}

// out = o2 @ w_o3^T + b_o3, split into (out[:, :64], out[:, 64:]) layout
__global__ void out_kernel(const float* __restrict__ o2,
                           const float* __restrict__ w,
                           const float* __restrict__ b,
                           float* __restrict__ out, int N) {
    __shared__ float lw[128 * 32];
    __shared__ float lx[2][32];
    int t = threadIdx.x;
    for (int i = t; i < 128 * 32; i += 256) lw[i] = w[i];
    __syncthreads();
    for (int nb = blockIdx.x * 2; nb < N; nb += gridDim.x * 2) {
        __syncthreads();
        if (t < 64) {
            int r = t >> 5, k = t & 31;
            int n = nb + r;
            lx[r][k] = (n < N) ? o2[(size_t)n * 32 + k] : 0.f;
        }
        __syncthreads();
        int r = t >> 7, j = t & 127;
        int n = nb + r;
        if (n < N) {
            float acc = b[j];
            for (int k = 0; k < 32; ++k) acc += lx[r][k] * lw[j * 32 + k];
            size_t o = (j < 64) ? ((size_t)n * 64 + j)
                                : ((size_t)N * 64 + (size_t)n * 64 + (j - 64));
            out[o] = acc;
        }
    }
}

// ---------------------------------------------------------------------------
extern "C" void kernel_launch(void* const* d_in, const int* in_sizes, int n_in,
                              void* d_out, int out_size, void* d_ws, size_t ws_size,
                              hipStream_t stream) {
    const float* x         = (const float*)d_in[0];
    const int*   node_type = (const int*)d_in[1];
    const int*   edge_type = (const int*)d_in[2];
    const int*   edge_index= (const int*)d_in[3];
    const int*   batch     = (const int*)d_in[4];
    const float* node_emb  = (const float*)d_in[5];
    const float* edge_emb  = (const float*)d_in[6];
    const float* w_d1 = (const float*)d_in[7];
    const float* g_d1 = (const float*)d_in[9];
    const float* be_d1= (const float*)d_in[10];
    const float* w_d2 = (const float*)d_in[11];
    const float* b_d2 = (const float*)d_in[12];
    const float* g_d2 = (const float*)d_in[13];
    const float* be_d2= (const float*)d_in[14];
    const float* g_c1 = (const float*)d_in[15];
    const float* be_c1= (const float*)d_in[16];
    const float* g_c2 = (const float*)d_in[17];
    const float* be_c2= (const float*)d_in[18];
    const float* g_c3 = (const float*)d_in[19];
    const float* be_c3= (const float*)d_in[20];
    const float* w_o1 = (const float*)d_in[21];
    const float* b_o1 = (const float*)d_in[22];
    const float* g_o1 = (const float*)d_in[23];
    const float* be_o1= (const float*)d_in[24];
    const float* w_o2 = (const float*)d_in[25];
    const float* b_o2 = (const float*)d_in[26];
    const float* g_o2 = (const float*)d_in[27];
    const float* be_o2= (const float*)d_in[28];
    const float* w_o3 = (const float*)d_in[29];
    const float* b_o3 = (const float*)d_in[30];

    const int E = in_sizes[0];      // 1,250,000
    const int N = in_sizes[1];      // 50,000
    const int* src = edge_index;
    const int* dst = edge_index + E;
    const size_t NH = (size_t)N * 64;

    // workspace layout (floats)
    float* wsf     = (float*)d_ws;
    float* xstats  = wsf;            // 2
    float* d2stats = wsf + 64;       // 128
    float* bnstats = wsf + 256;      // 128
    float* hg      = wsf + 1024;     // BATCHES*64 = 32000
    float* hA      = wsf + 40960;    // N*64
    float* hB      = hA + NH + 1024; // agg / pre-activation
    float* hC      = hB + NH + 1024;
    float* o2buf   = hC + NH + 1024; // N*32
    _Float16* ea   = (_Float16*)(o2buf + (size_t)N * 32 + 1024); // E*64 f16

    // --- edge feature pipeline -------------------------------------------
    zero_kernel<<<1, 256, 0, stream>>>(wsf, 1024);               // all stats
    zero_kernel<<<32, 256, 0, stream>>>(hg, (long)BATCHES * 64);
    reduce_x_kernel<<<512, 256, 0, stream>>>(x, xstats, E);

    int gblocks = (E + 31) / 32;
    gemm_d2_kernel<<<gblocks, 256, 0, stream>>>(x, w_d1, g_d1, be_d1,
                                                w_d2, b_d2, xstats, d2stats,
                                                ea, E);
    bn2_ea_kernel<<<4096, 256, 0, stream>>>(ea, d2stats, g_d2, be_d2,
                                            edge_emb, edge_type, E);

    // --- node init + 3 gconv layers --------------------------------------
    gather_nodes_kernel<<<1024, 256, 0, stream>>>(node_emb, node_type, hA, N);

    float* cur = hA;
    float* nxt = hC;
    const float* gl[3] = {g_c1, g_c2, g_c3};
    const float* bl[3] = {be_c1, be_c2, be_c3};
    for (int l = 0; l < 3; ++l) {
        copy_kernel<<<1024, 256, 0, stream>>>(hB, cur, (long)NH);
        zero_kernel<<<1, 256, 0, stream>>>(bnstats, 128);
        gconv_scatter_kernel<<<4096, 256, 0, stream>>>(cur, ea, src, dst, hB, E);
        colstats_kernel<<<512, 256, 0, stream>>>(hB, bnstats, N, 64);
        bn_apply_kernel<<<2048, 256, 0, stream>>>(hB, nxt, bnstats, gl[l], bl[l],
                                                  N, 64, (l < 2) ? 1 : 0);
        float* tmp = cur; cur = nxt; nxt = tmp;
    }
    // after loop: cur = h3, nxt = free buffer

    // --- global pooling + output head ------------------------------------
    hg_scatter_kernel<<<1024, 256, 0, stream>>>(cur, batch, hg, N);

    o1_kernel<<<4096, 256, 0, stream>>>(cur, hg, batch, w_o1, b_o1, hB, N);
    zero_kernel<<<1, 256, 0, stream>>>(bnstats, 128);
    colstats_kernel<<<512, 256, 0, stream>>>(hB, bnstats, N, 64);
    bn_apply_kernel<<<2048, 256, 0, stream>>>(hB, nxt, bnstats, g_o1, be_o1,
                                              N, 64, 1);

    o2_kernel<<<2048, 256, 0, stream>>>(nxt, w_o2, b_o2, o2buf, N);
    zero_kernel<<<1, 256, 0, stream>>>(bnstats, 128);
    colstats_kernel<<<512, 256, 0, stream>>>(o2buf, bnstats, N, 32);
    bn_apply_kernel<<<1024, 256, 0, stream>>>(o2buf, o2buf, bnstats, g_o2, be_o2,
                                              N, 32, 1);

    out_kernel<<<4096, 256, 0, stream>>>(o2buf, w_o3, b_o3, (float*)d_out, N);

    (void)n_in; (void)out_size; (void)ws_size;
}
